// RelAwareRelFeature_31181462569561
// MI455X (gfx1250) — matrix operand order repaired and checked by reading, not compiled, not run on hardware
//
#include <hip/hip_runtime.h>
#include <hip/hip_bf16.h>
#include <math.h>

typedef __attribute__((ext_vector_type(16))) _Float16 v16h;
typedef __attribute__((ext_vector_type(8)))  _Float16 v8h;
typedef __attribute__((ext_vector_type(4)))  _Float16 v4h;
typedef __attribute__((ext_vector_type(8)))  float    v8f;
typedef __attribute__((ext_vector_type(4)))  float    v4f;

#define P_PAIRS 200000
#define N_BOX   1000
#define D_IN    512
#define E_SEM   300
#define E_PAD   320
#define G_POS   128
#define H_DIM   512
#define C_RELN  50
#define KGEO    896      // 128 + 320 + 128 + 320 (each sem segment zero-padded)
#define FUSED   1024

// workspace layout in halves (all offsets multiples of 16 halves = 32B)
#define OFF_RELUPOS 0
#define OFF_RELUSEM (OFF_RELUPOS + N_BOX*G_POS)          // 128000
#define OFF_VISPACK (OFF_RELUSEM + N_BOX*E_PAD)          // +320000
#define OFF_GEOPACK (OFF_VISPACK + 16*32*512)            // +262144
#define OFF_FUSPACK (OFF_GEOPACK + 28*32*512)            // +458752
#define OFF_CLSPACK (OFF_FUSPACK + 32*32*512)            // +524288

// ---------------- prep: pos MLP (N=1000 boxes) -> relu(pos) f16 -------------
__global__ void prep_pos(const float* __restrict__ bbox,
                         const float* __restrict__ w1, const float* __restrict__ b1,
                         const float* __restrict__ w2, const float* __restrict__ b2,
                         _Float16* __restrict__ reluPos) {
  __shared__ float h1[G_POS];
  __shared__ float f[9];
  int i = blockIdx.x, j = threadIdx.x;
  if (j == 0) {
    float x1 = bbox[i*4+0], y1 = bbox[i*4+1], x2 = bbox[i*4+2], y2 = bbox[i*4+3];
    float w = x2 - x1 + 1.f, h = y2 - y1 + 1.f;
    f[0]=w; f[1]=h; f[2]=x1+0.5f*w; f[3]=y1+0.5f*h;
    f[4]=x1; f[5]=y1; f[6]=x2; f[7]=y2; f[8]=w*h;
  }
  __syncthreads();
  float s = b1[j];
#pragma unroll
  for (int k = 0; k < 9; ++k) s += f[k]*w1[k*G_POS + j];
  h1[j] = s > 0.f ? s : 0.f;
  __syncthreads();
  float s2 = b2[j];
  for (int k = 0; k < G_POS; ++k) s2 += h1[k]*w2[k*G_POS + j];
  reluPos[i*G_POS + j] = (_Float16)(s2 > 0.f ? s2 : 0.f);
}

// ---------------- prep: relu(sem gather) f16, padded to 320 -----------------
__global__ void prep_sem(const float* __restrict__ sem_w, const int* __restrict__ labels,
                         _Float16* __restrict__ reluSem) {
  int idx = blockIdx.x*256 + threadIdx.x;
  if (idx >= N_BOX*E_PAD) return;
  int i = idx / E_PAD, j = idx - i*E_PAD;
  float v = 0.f;
  if (j < E_SEM) {
    v = sem_w[(size_t)labels[i]*E_SEM + j];
    v = v > 0.f ? v : 0.f;
  }
  reluSem[idx] = (_Float16)v;
}

// ---------------- pack weights into WMMA B-fragment order -------------------
// dst layout: [ks][tile][lane][h]; B 32x16 f16: lane L holds N=L%16, K = (h&7) + 8*(L/16) + 16*(h/8)
__global__ void pack_weights(const float* __restrict__ W, _Float16* __restrict__ dst,
                             int ksteps, int ntiles, int Korig, int Norig, int geo_remap) {
  int idx = blockIdx.x*256 + threadIdx.x;
  int total = ksteps*ntiles*512;
  if (idx >= total) return;
  int h    = idx & 15;
  int lane = (idx >> 4) & 31;
  int tile = (idx >> 9) % ntiles;
  int ks   = (idx >> 9) / ntiles;
  int K = ks*32 + (h & 7) + ((lane >> 4) << 3) + ((h >> 3) << 4);
  int N = tile*16 + (lane & 15);
  int kk = K, valid = 1;
  if (geo_remap) {              // padded K in [0,896) -> geo_w row in [0,856)
    if      (K < 128) { kk = K; }
    else if (K < 448) { int c = K-128; kk = 128 + c; valid = (c < E_SEM); }
    else if (K < 576) { kk = 428 + (K-448); }
    else              { int c = K-576; kk = 556 + c; valid = (c < E_SEM); }
  } else {
    valid = (K < Korig);
  }
  float val = 0.f;
  if (valid && N < Norig) val = W[(size_t)kk*Norig + N];
  dst[idx] = (_Float16)val;
}

__global__ void zero_f32(float* p, int n) {
  int i = blockIdx.x*256 + threadIdx.x;
  if (i < n) p[i] = 0.f;
}

// ---------------- fragment loaders ------------------------------------------
// A 16x32 f16 from row-major LDS tile. ISA layout: lanes 0-15 (M=lane):
//   halves 0..7 = K 0..7, halves 8..15 = K 16..23; lanes 16-31 shifted +8.
__device__ __forceinline__ v16h load_a_frag(const _Float16* __restrict__ buf,
                                            int strideH, int lane, int ks) {
  int m = lane & 15;
  int kbase = ks*32 + ((lane >> 4) << 3);
  const _Float16* p = buf + m*strideH + kbase;   // 16B aligned for our strides
  v8h lo = *(const v8h*)(p);
  v8h hi = *(const v8h*)(p + 16);
  v16h a;
#pragma unroll
  for (int i = 0; i < 8; ++i) { a[i] = lo[i]; a[i+8] = hi[i]; }
  return a;
}

__device__ __forceinline__ v16h load_b_frag(const _Float16* __restrict__ pack,
                                            int ntiles, int ks, int tile, int lane) {
  return *(const v16h*)(pack + ((((size_t)ks*ntiles + tile)*32) + lane)*16);
}

__device__ __forceinline__ void run_gemm(const _Float16* __restrict__ aBuf, int strideH,
                                         const _Float16* __restrict__ pack, int ntilesTot,
                                         int ksteps, int wave, int lane, v8f acc[4]) {
  for (int ks = 0; ks < ksteps; ++ks) {
    v16h a = load_a_frag(aBuf, strideH, lane, ks);
#pragma unroll
    for (int t = 0; t < 4; ++t) {
      v16h b = load_b_frag(pack, ntilesTot, ks, wave*4 + t, lane);
      acc[t] = __builtin_amdgcn_wmma_f32_16x16x32_f16(false, a, false, b,
                                                      (short)0, acc[t], false, false);
    }
  }
}

// C/D layout: VGPR r -> M = r + 8*(lane/16); N = lane%16 within tile
__device__ __forceinline__ void store_c(float* __restrict__ dst, int strideF, int colOff,
                                        const float* __restrict__ bias,
                                        int wave, int lane, const v8f acc[4]) {
  int rbase = (lane >> 4) << 3;
  int nl = lane & 15;
#pragma unroll
  for (int t = 0; t < 4; ++t) {
    int col = wave*64 + t*16 + nl;
    float bv = bias[col];
#pragma unroll
    for (int r = 0; r < 8; ++r)
      dst[(rbase + r)*strideF + colOff + col] = acc[t][r] + bv;
  }
}

__device__ __forceinline__ void ln_relu(const float* __restrict__ src,
                                        _Float16* __restrict__ dst, int W,
                                        const float* __restrict__ g,
                                        const float* __restrict__ b,
                                        int wave, int lane) {
#pragma unroll
  for (int rr = 0; rr < 2; ++rr) {
    int row = wave*2 + rr;
    float s = 0.f, s2 = 0.f;
    for (int c = lane; c < W; c += 32) { float v = src[row*W + c]; s += v; s2 += v*v; }
#pragma unroll
    for (int off = 16; off > 0; off >>= 1) {
      s  += __shfl_xor(s,  off, 32);
      s2 += __shfl_xor(s2, off, 32);
    }
    float mean = s / (float)W;
    float var  = s2 / (float)W - mean*mean;
    float rstd = rsqrtf(var + 1e-5f);
    for (int c = lane; c < W; c += 32) {
      float v = (src[row*W + c] - mean)*rstd*g[c] + b[c];
      dst[row*W + c] = (_Float16)(v > 0.f ? v : 0.f);
    }
  }
}

// ---------------- mega-fused main kernel: 16 pair-rows per block ------------
__global__ __launch_bounds__(256) void fused_main(
    const float* __restrict__ visual, const int* __restrict__ pair_idx,
    const _Float16* __restrict__ ws,
    const float* __restrict__ vis_b, const float* __restrict__ geo_b,
    const float* __restrict__ fln_g, const float* __restrict__ fln_b,
    const float* __restrict__ fus_b,
    const float* __restrict__ cln_g, const float* __restrict__ cln_b,
    const float* __restrict__ cls_b,
    float* __restrict__ out_logits, float* __restrict__ out_rel) {
  __shared__ __align__(16) _Float16 aF16[16*FUSED];   // 32 KB: f16 A tiles
  __shared__ __align__(16) float    accS[16*FUSED];   // 64 KB: f32 results
  __shared__ int sIdx[16], oIdx[16];

  const int tid = threadIdx.x, lane = tid & 31, wave = tid >> 5;
  const int p0 = blockIdx.x * 16;

  const _Float16* reluPos = ws + OFF_RELUPOS;
  const _Float16* reluSem = ws + OFF_RELUSEM;
  const _Float16* visPack = ws + OFF_VISPACK;
  const _Float16* geoPack = ws + OFF_GEOPACK;
  const _Float16* fusPack = ws + OFF_FUSPACK;
  const _Float16* clsPack = ws + OFF_CLSPACK;

  if (tid < 16) {
    sIdx[tid] = pair_idx[(p0 + tid)*2 + 0];
    oIdx[tid] = pair_idx[(p0 + tid)*2 + 1];
  }
  // ---- stage 1: vis A tile (16x512 f32 -> f16), vectorized b128 loads ------
  {
    const v4f* src = (const v4f*)(visual + (size_t)p0*D_IN);
#pragma unroll
    for (int it = 0; it < (16*D_IN/4)/256; ++it) {      // 8 iterations
      int idx = it*256 + tid;                           // float4 chunk index
      v4f x = src[idx];
      v4h y;
#pragma unroll
      for (int i = 0; i < 4; ++i) y[i] = (_Float16)x[i];
      *(v4h*)(aF16 + idx*4) = y;
    }
  }
  __syncthreads();

  v8f acc4[4];
#pragma unroll
  for (int t = 0; t < 4; ++t)
#pragma unroll
    for (int i = 0; i < 8; ++i) acc4[t][i] = 0.f;

  run_gemm(aF16, D_IN, visPack, 32, 16, wave, lane, acc4);     // vis GEMM
  store_c(accS, FUSED, 0, vis_b, wave, lane, acc4);
  __syncthreads();

  // ---- stage 2: geo A tile gather in 16-byte chunks (16x896 f16) -----------
  for (int idx = tid; idx < 16*(KGEO/8); idx += 256) {  // 1792 v8h chunks
    int r  = idx / (KGEO/8);
    int c8 = (idx - r*(KGEO/8)) * 8;
    const _Float16* srcp;
    if      (c8 < 128) srcp = reluPos + sIdx[r]*G_POS + c8;
    else if (c8 < 448) srcp = reluSem + sIdx[r]*E_PAD + (c8 - 128);
    else if (c8 < 576) srcp = reluPos + oIdx[r]*G_POS + (c8 - 448);
    else               srcp = reluSem + oIdx[r]*E_PAD + (c8 - 576);
    *(v8h*)(aF16 + r*KGEO + c8) = *(const v8h*)srcp;
  }
  __syncthreads();

#pragma unroll
  for (int t = 0; t < 4; ++t)
#pragma unroll
    for (int i = 0; i < 8; ++i) acc4[t][i] = 0.f;
  run_gemm(aF16, KGEO, geoPack, 32, 28, wave, lane, acc4);     // geo GEMM
  store_c(accS, FUSED, 512, geo_b, wave, lane, acc4);
  __syncthreads();

  // ---- stage 3: layernorm(1024)+relu -> f16 A ------------------------------
  ln_relu(accS, aF16, FUSED, fln_g, fln_b, wave, lane);
  __syncthreads();

#pragma unroll
  for (int t = 0; t < 4; ++t)
#pragma unroll
    for (int i = 0; i < 8; ++i) acc4[t][i] = 0.f;
  run_gemm(aF16, FUSED, fusPack, 32, 32, wave, lane, acc4);    // fus GEMM
  store_c(accS, H_DIM, 0, fus_b, wave, lane, acc4);
  __syncthreads();

  // ---- stage 4: layernorm(512)+relu -> f16 A -------------------------------
  ln_relu(accS, aF16, H_DIM, cln_g, cln_b, wave, lane);
  __syncthreads();

  // ---- stage 5: cls GEMM (512 -> 64 padded cols, waves 0-3) ----------------
  if (wave < 4) {
    v8f c4;
#pragma unroll
    for (int i = 0; i < 8; ++i) c4[i] = 0.f;
    for (int ks = 0; ks < 16; ++ks) {
      v16h a = load_a_frag(aF16, H_DIM, lane, ks);
      v16h b = load_b_frag(clsPack, 4, ks, wave, lane);
      c4 = __builtin_amdgcn_wmma_f32_16x16x32_f16(false, a, false, b,
                                                  (short)0, c4, false, false);
    }
    int col = wave*16 + (lane & 15);
    int rbase = (lane >> 4) << 3;
#pragma unroll
    for (int r = 0; r < 8; ++r) {
      int row = rbase + r;
      if (col < C_RELN) {
        float v = c4[r] + cls_b[col];
        out_logits[(size_t)(p0 + row)*C_RELN + col] = v;
        accS[row*64 + col] = 1.f/(1.f + __expf(-v));      // sigmoid for max
      }
    }
  }
  __syncthreads();
  if (tid < 16) {
    float mx = 0.f;
    for (int j = 0; j < C_RELN; ++j) mx = fmaxf(mx, accS[tid*64 + j]);
    out_rel[(size_t)sIdx[tid]*N_BOX + oIdx[tid]] = mx;
  }
}

// ---------------------------------------------------------------------------
extern "C" void kernel_launch(void* const* d_in, const int* in_sizes, int n_in,
                              void* d_out, int out_size, void* d_ws, size_t ws_size,
                              hipStream_t stream) {
  const float* visual = (const float*)d_in[0];
  const float* bbox   = (const float*)d_in[1];
  const int*   labels = (const int*)  d_in[2];
  const int*   pair   = (const int*)  d_in[3];
  const float* sem_w  = (const float*)d_in[4];
  const float* pw1 = (const float*)d_in[5],  *pb1 = (const float*)d_in[6];
  const float* pw2 = (const float*)d_in[7],  *pb2 = (const float*)d_in[8];
  const float* vis_w = (const float*)d_in[9],  *vis_b = (const float*)d_in[10];
  const float* geo_w = (const float*)d_in[11], *geo_b = (const float*)d_in[12];
  const float* fln_g = (const float*)d_in[13], *fln_b = (const float*)d_in[14];
  const float* fus_w = (const float*)d_in[15], *fus_b = (const float*)d_in[16];
  const float* cln_g = (const float*)d_in[17], *cln_b = (const float*)d_in[18];
  const float* cls_w = (const float*)d_in[19], *cls_b = (const float*)d_in[20];

  _Float16* ws = (_Float16*)d_ws;
  float* out_logits = (float*)d_out;
  float* out_rel    = out_logits + (size_t)P_PAIRS*C_RELN;

  prep_pos<<<N_BOX, G_POS, 0, stream>>>(bbox, pw1, pb1, pw2, pb2, ws + OFF_RELUPOS);
  prep_sem<<<(N_BOX*E_PAD + 255)/256, 256, 0, stream>>>(sem_w, labels, ws + OFF_RELUSEM);

  pack_weights<<<(16*32*512 + 255)/256, 256, 0, stream>>>(vis_w, ws + OFF_VISPACK, 16, 32, 512, 512, 0);
  pack_weights<<<(28*32*512 + 255)/256, 256, 0, stream>>>(geo_w, ws + OFF_GEOPACK, 28, 32, 856, 512, 1);
  pack_weights<<<(32*32*512 + 255)/256, 256, 0, stream>>>(fus_w, ws + OFF_FUSPACK, 32, 32, 1024, 512, 0);
  pack_weights<<<(16*4*512  + 255)/256, 256, 0, stream>>>(cls_w, ws + OFF_CLSPACK, 16, 4, 512, 50, 0);

  zero_f32<<<(N_BOX*N_BOX + 255)/256, 256, 0, stream>>>(out_rel, N_BOX*N_BOX);

  fused_main<<<P_PAIRS/16, 256, 0, stream>>>(visual, pair, ws,
      vis_b, geo_b, fln_g, fln_b, fus_b, cln_g, cln_b, cls_b,
      out_logits, out_rel);
}
